// AdaptiveAttentionTransformerBlock_34179349741877
// MI455X (gfx1250) — matrix operand order repaired and checked by reading, not compiled, not run on hardware
//
#include <hip/hip_runtime.h>
#include <math.h>

typedef __attribute__((ext_vector_type(16))) __bf16 v16bf;
typedef __attribute__((ext_vector_type(8)))  float  v8f;

union BF16Frag {
  v16bf v;
  uint4 q[2];
  unsigned short u[16];
};

union U16x16 { uint4 q[2]; unsigned short u[16]; };

__device__ __forceinline__ unsigned short f2bf(float f) {
  return __builtin_bit_cast(unsigned short, (__bf16)f);   // native cvt, RNE
}
__device__ __forceinline__ float bf2f(unsigned short u) {
  return __uint_as_float((unsigned)u << 16);
}
__device__ __forceinline__ v8f v8f_zero() {
  v8f z = {0.f,0.f,0.f,0.f,0.f,0.f,0.f,0.f};
  return z;
}

// Async DMA: global -> LDS, 16 bytes per lane (ASYNCcnt-tracked).
__device__ __forceinline__ void async_b128(void* lds, const void* gptr) {
  unsigned lds_off = (unsigned)(size_t)lds;   // low 32 bits of generic = LDS offset
  asm volatile("global_load_async_to_lds_b128 %0, %1, off"
               :: "v"(lds_off), "v"(gptr) : "memory");
}
__device__ __forceinline__ void wait_async0() {
  asm volatile("s_wait_asynccnt 0" ::: "memory");
}

// ---- DPP 16-lane butterfly reductions (no LDS traffic) ---------------------
constexpr unsigned dpp8_sel_xor(int m) {
  unsigned v = 0;
  for (int j = 0; j < 8; ++j) v |= (unsigned)(j ^ m) << (3 * j);
  return v;
}
template <unsigned SEL>
__device__ __forceinline__ float mov_dpp8_f(float x) {
  return __builtin_bit_cast(float,
      __builtin_amdgcn_mov_dpp8(__builtin_bit_cast(int, x), SEL));
}
__device__ __forceinline__ float ror8_row(float x) {   // xor-8 within 16-lane row
  int i = __builtin_bit_cast(int, x);
  int r = __builtin_amdgcn_update_dpp(i, i, 0x128 /*row_ror:8*/, 0xF, 0xF, true);
  return __builtin_bit_cast(float, r);
}
__device__ __forceinline__ float rowmax16(float x) {
  x = fmaxf(x, mov_dpp8_f<dpp8_sel_xor(1)>(x));
  x = fmaxf(x, mov_dpp8_f<dpp8_sel_xor(2)>(x));
  x = fmaxf(x, mov_dpp8_f<dpp8_sel_xor(4)>(x));
  x = fmaxf(x, ror8_row(x));
  return x;
}
__device__ __forceinline__ float rowsum16(float x) {
  x += mov_dpp8_f<dpp8_sel_xor(1)>(x);
  x += mov_dpp8_f<dpp8_sel_xor(2)>(x);
  x += mov_dpp8_f<dpp8_sel_xor(4)>(x);
  x += ror8_row(x);
  return x;
}

// ---------------------------------------------------------------------------
__global__ void f32_to_bf16(const float* __restrict__ s,
                            unsigned short* __restrict__ d, int n) {
  int i = blockIdx.x * 256 + threadIdx.x;
  if (i < n) d[i] = f2bf(s[i]);
}

// ---------------------------------------------------------------------------
// C[M][N] = A[M][K] * B[N][K]^T ; A,B bf16 in HBM. Double-buffered LDS:
// next tile DMA'd (A, async) / transpose-scattered (B) while WMMAs run on
// the current tile. Block tile 256x128, K-step 32; 8 waves as 4(M) x 2(N),
// wave tile 64x64 -> 16 WMMAs per 8 ds_load_b128 (A) + 64 u16 LDS reads (B) per K-step.
// ---------------------------------------------------------------------------
template <bool STORE_BF16>
__global__ __launch_bounds__(256) void gemm_bf16_tn(
    const unsigned short* __restrict__ A, const unsigned short* __restrict__ B,
    void* __restrict__ Cv, int M, int N, int K)
{
  __shared__ unsigned short As[2][256][32];    // [buf][m][k]
  __shared__ unsigned short Bst[2][32][136];   // [buf][k][n] (+8 pad)

  const int tid  = threadIdx.x;
  const int lane = tid & 31;
  const int wid  = tid >> 5;
  const int g    = lane >> 4;
  const int mr   = lane & 15;
  const int wm   = wid & 3;     // 4 waves along M
  const int wn   = wid >> 2;    // 2 waves along N
  const int m0   = blockIdx.y * 256;
  const int n0   = blockIdx.x * 128;

  v8f acc[4][4];
#pragma unroll
  for (int mt = 0; mt < 4; ++mt)
#pragma unroll
    for (int nt = 0; nt < 4; ++nt) acc[mt][nt] = v8f_zero();

  const int arow = tid;              // A stage: one 32-elem row per thread
  const int brow = tid >> 1;         // B stage: half row per thread
  const int bcs  = (tid & 1) * 16;

  auto stage = [&](int buf, int k0) {
    const unsigned short* gA = A + (size_t)(m0 + arow) * K + k0;
    async_b128(&As[buf][arow][0],  gA);
    async_b128(&As[buf][arow][8],  gA + 8);
    async_b128(&As[buf][arow][16], gA + 16);
    async_b128(&As[buf][arow][24], gA + 24);
    const uint4* gb4 = (const uint4*)(B + (size_t)(n0 + brow) * K + k0 + bcs);
    U16x16 tb; tb.q[0] = gb4[0]; tb.q[1] = gb4[1];
#pragma unroll
    for (int e = 0; e < 16; ++e) Bst[buf][bcs + e][brow] = tb.u[e];
  };

  stage(0, 0);
  wait_async0();
  __syncthreads();

  const int NK = K >> 5;
  for (int kt = 0; kt < NK; ++kt) {
    const int cur = kt & 1;
    if (kt + 1 < NK) stage(cur ^ 1, (kt + 1) * 32);

    BF16Frag afr[4];
#pragma unroll
    for (int mt = 0; mt < 4; ++mt) {
      const int row = wm * 64 + mt * 16 + mr;
      afr[mt].q[0] = *(const uint4*)&As[cur][row][g * 8];
      afr[mt].q[1] = *(const uint4*)&As[cur][row][16 + g * 8];
    }
#pragma unroll
    for (int nt = 0; nt < 4; ++nt) {
      BF16Frag bfr;
      const int cb = wn * 64 + nt * 16;
      for (int e = 0; e < 16; ++e)
        bfr.u[e] = Bst[cur][(e & 7) + 8 * g + (e >> 3) * 16][cb + mr];
#pragma unroll
      for (int mt = 0; mt < 4; ++mt)
        acc[mt][nt] = __builtin_amdgcn_wmma_f32_16x16x32_bf16(
            false, afr[mt].v, false, bfr.v, (short)0, acc[mt][nt], false, false);
    }
    wait_async0();
    __syncthreads();
  }

#pragma unroll
  for (int mt = 0; mt < 4; ++mt)
#pragma unroll
    for (int nt = 0; nt < 4; ++nt)
#pragma unroll
      for (int r = 0; r < 8; ++r) {
        const int row = m0 + wm * 64 + mt * 16 + r + 8 * g;
        const int col = n0 + wn * 64 + nt * 16 + mr;
        if (STORE_BF16)
          ((unsigned short*)Cv)[(size_t)row * N + col] = f2bf(acc[mt][nt][r]);
        else
          ((float*)Cv)[(size_t)row * N + col] = acc[mt][nt][r];
      }
}

// ---------------------------------------------------------------------------
// RoPE in-place on bf16 qkv[4096][3072] (q and k slices).
// ---------------------------------------------------------------------------
__global__ void rope_inplace_bf16(unsigned short* __restrict__ qkv)
{
  const int idx = blockIdx.x * blockDim.x + threadIdx.x;  // B*S*H*32
  const int pr = idx & 31;
  const int h  = (idx >> 5) & 15;
  const int t  = idx >> 9;          // token 0..4095
  const int s  = t & 2047;
  unsigned short* qp = qkv + (size_t)t * 3072 + h * 64;
  const float LOGB_32 = 0.2878231366242557f;  // ln(10000)/32
  const int d0 = pr * 2, d1 = d0 + 1;
  const float th0 = (float)s * __expf(-(float)(d0 & 31) * LOGB_32);
  const float th1 = (float)s * __expf(-(float)(d1 & 31) * LOGB_32);
  float c0, s0, c1, s1;
  __sincosf(th0, &s0, &c0);
  __sincosf(th1, &s1, &c1);
  const float q0 = bf2f(qp[d0]), q1 = bf2f(qp[d1]);
  qp[d0] = f2bf(q0 * c0 - q1 * s0);
  qp[d1] = f2bf(q1 * c1 + q0 * s1);
  unsigned short* kp = qp + 1024;
  const float k0 = bf2f(kp[d0]), k1 = bf2f(kp[d1]);
  kp[d0] = f2bf(k0 * c0 - k1 * s0);
  kp[d1] = f2bf(k1 * c1 + k0 * s1);
}

// ---------------------------------------------------------------------------
// Block-sparse flash attention on bf16 qkv. Grid (S/128, B*H), 8 waves.
// q/v tiles via async DMA, k tile transposed; k/v double-buffered across the
// key-block list. Softmax row reductions via DPP (no LDS).
// ---------------------------------------------------------------------------
__global__ __launch_bounds__(256) void attn_sparse(
    const unsigned short* __restrict__ qkv,   // [4096][3072] bf16
    unsigned short* __restrict__ aout)        // [4096][1024] bf16
{
  __shared__ unsigned short qs[128][64];        // [query][dim]
  __shared__ unsigned short kst[2][64][72];     // [buf][dim][key] (+8 pad)
  __shared__ unsigned short vs[2][64][64];      // [buf][key][dim]
  __shared__ unsigned short ps[8][16 * 64];     // per-wave P staging

  const int tid  = threadIdx.x;
  const int lane = tid & 31;
  const int wid  = tid >> 5;
  const int g    = lane >> 4;
  const int mr   = lane & 15;
  const int qb   = blockIdx.x;
  const int q0   = qb * 128;
  const int b    = blockIdx.y >> 4;
  const int h    = blockIdx.y & 15;
  const unsigned short* base = qkv + (size_t)b * 2048 * 3072 + h * 64;

  auto stage_kv = [&](int buf, int k0) {
    for (int i = tid; i < 64 * 64; i += 256) {         // k: transpose scatter
      const int r = i >> 6, c = i & 63;
      kst[buf][c][r] = base[(size_t)(k0 + r) * 3072 + 1024 + c];
    }
    for (int c2 = tid; c2 < 512; c2 += 256) {          // v: async DMA
      const int r = c2 >> 3, c8 = (c2 & 7) * 8;
      async_b128(&vs[buf][r][c8], base + (size_t)(k0 + r) * 3072 + 2048 + c8);
    }
  };

  // q tile: 128 rows x 128B -> 1024 async 16B chunks
  for (int c2 = tid; c2 < 1024; c2 += 256) {
    const int r = c2 >> 3, c8 = (c2 & 7) * 8;
    async_b128(&qs[r][c8], base + (size_t)(q0 + r) * 3072 + c8);
  }

  int list[5]; int nkb = 0;
  int first = qb * 2 - 2; if (first < 0) first = 0;
  if (first > 0) list[nkb++] = 0;                 // global-token block
  for (int kb = first; kb <= qb * 2 + 1; ++kb) list[nkb++] = kb;

  stage_kv(0, list[0] * 64);
  wait_async0();
  __syncthreads();

  const int qrl = wid * 16;
  BF16Frag aq[2];
#pragma unroll
  for (int kk = 0; kk < 2; ++kk) {
    aq[kk].q[0] = *(const uint4*)&qs[qrl + mr][kk * 32 + g * 8];
    aq[kk].q[1] = *(const uint4*)&qs[qrl + mr][kk * 32 + 16 + g * 8];
  }

  float mrow[8], lsum[8];
  v8f oacc[4];
#pragma unroll
  for (int r = 0; r < 8; ++r) { mrow[r] = -1e30f; lsum[r] = 0.f; }
#pragma unroll
  for (int n = 0; n < 4; ++n) oacc[n] = v8f_zero();

  const float scale = 0.125f;  // 64^-0.5
  for (int ik = 0; ik < nkb; ++ik) {
    const int cur = ik & 1;
    if (ik + 1 < nkb) stage_kv(cur ^ 1, list[ik + 1] * 64);
    const int k0 = list[ik] * 64;

    float st[4][8];
#pragma unroll
    for (int ct = 0; ct < 4; ++ct) {
      v8f s = v8f_zero();
#pragma unroll
      for (int kk = 0; kk < 2; ++kk) {
        BF16Frag bk;
        for (int e = 0; e < 16; ++e)
          bk.u[e] = kst[cur][kk * 32 + (e & 7) + 8 * g + (e >> 3) * 16][ct * 16 + mr];
        s = __builtin_amdgcn_wmma_f32_16x16x32_bf16(
            false, aq[kk].v, false, bk.v, (short)0, s, false, false);
      }
#pragma unroll
      for (int r = 0; r < 8; ++r) {
        const int i = q0 + qrl + r + 8 * g;
        const int j = k0 + ct * 16 + mr;
        const bool ok = (j <= i) &&
            (((i >> 8) == (j >> 8)) || (i < 4) || (j < 4) || ((i - j) <= 128));
        st[ct][r] = ok ? s[r] * scale : -1e30f;
      }
    }
#pragma unroll
    for (int r = 0; r < 8; ++r) {
      float mx = st[0][r];
#pragma unroll
      for (int ct = 1; ct < 4; ++ct) mx = fmaxf(mx, st[ct][r]);
      mx = rowmax16(mx);
      const float mnew  = fmaxf(mrow[r], mx);
      const float alpha = __expf(mrow[r] - mnew);
      mrow[r] = mnew;
      lsum[r] *= alpha;
#pragma unroll
      for (int n = 0; n < 4; ++n) oacc[n][r] *= alpha;
    }
#pragma unroll
    for (int ct = 0; ct < 4; ++ct)
#pragma unroll
      for (int r = 0; r < 8; ++r) {
        const float p = __expf(st[ct][r] - mrow[r]);
        st[ct][r] = p;
        ps[wid][(r + 8 * g) * 64 + ct * 16 + mr] = f2bf(p);
      }
#pragma unroll
    for (int r = 0; r < 8; ++r)
      lsum[r] += rowsum16(st[0][r] + st[1][r] + st[2][r] + st[3][r]);

    // P (16x64) @ V (64x64)
#pragma unroll
    for (int kk = 0; kk < 2; ++kk) {
      BF16Frag ap;
      ap.q[0] = *(const uint4*)&ps[wid][mr * 64 + kk * 32 + g * 8];
      ap.q[1] = *(const uint4*)&ps[wid][mr * 64 + kk * 32 + 16 + g * 8];
#pragma unroll
      for (int n = 0; n < 4; ++n) {
        BF16Frag bv;
        for (int e = 0; e < 16; ++e)
          bv.u[e] = vs[cur][kk * 32 + (e & 7) + 8 * g + (e >> 3) * 16][n * 16 + mr];
        oacc[n] = __builtin_amdgcn_wmma_f32_16x16x32_bf16(
            false, ap.v, false, bv.v, (short)0, oacc[n], false, false);
      }
    }
    wait_async0();
    __syncthreads();
  }

#pragma unroll
  for (int n = 0; n < 4; ++n)
#pragma unroll
    for (int r = 0; r < 8; ++r) {
      const int t   = b * 2048 + q0 + qrl + r + 8 * g;
      const int col = h * 64 + n * 16 + mr;
      aout[(size_t)t * 1024 + col] = f2bf(oacc[n][r] / lsum[r]);
    }
}

// ---------------------------------------------------------------------------
extern "C" void kernel_launch(void* const* d_in, const int* in_sizes, int n_in,
                              void* d_out, int out_size, void* d_ws, size_t ws_size,
                              hipStream_t stream) {
  (void)in_sizes; (void)n_in; (void)out_size; (void)ws_size;
  const float* x     = (const float*)d_in[0];   // [2,2048,1024]
  const float* qkv_w = (const float*)d_in[1];   // [3072,1024]
  const float* out_w = (const float*)d_in[2];   // [1024,1024]
  float* out = (float*)d_out;                   // [2,2048,1024]

  char* ws = (char*)d_ws;
  size_t off = 0;
  unsigned short* xbf   = (unsigned short*)(ws + off); off += (size_t)4096 * 1024 * 2;
  unsigned short* wqkv  = (unsigned short*)(ws + off); off += (size_t)3072 * 1024 * 2;
  unsigned short* wout  = (unsigned short*)(ws + off); off += (size_t)1024 * 1024 * 2;
  unsigned short* qkvbf = (unsigned short*)(ws + off); off += (size_t)4096 * 3072 * 2;
  unsigned short* attnb = (unsigned short*)(ws + off); off += (size_t)4096 * 1024 * 2;

  // 0) one-time bf16 conversions (inputs stay untouched)
  f32_to_bf16<<<(4096 * 1024 + 255) / 256, 256, 0, stream>>>(x, xbf, 4096 * 1024);
  f32_to_bf16<<<(3072 * 1024 + 255) / 256, 256, 0, stream>>>(qkv_w, wqkv, 3072 * 1024);
  f32_to_bf16<<<(1024 * 1024 + 255) / 256, 256, 0, stream>>>(out_w, wout, 1024 * 1024);
  // 1) qkv = x @ qkv_w^T  -> bf16   (M=4096, N=3072, K=1024)
  gemm_bf16_tn<true><<<dim3(3072 / 128, 4096 / 256), 256, 0, stream>>>(
      xbf, wqkv, qkvbf, 4096, 3072, 1024);
  // 2) RoPE on q,k in place
  rope_inplace_bf16<<<(4096 * 16 * 32) / 256, 256, 0, stream>>>(qkvbf);
  // 3) block-sparse attention -> bf16 [4096][1024] (b,s,h,d order)
  attn_sparse<<<dim3(16, 32), 256, 0, stream>>>(qkvbf, attnb);
  // 4) out = attn @ out_w^T -> f32  (M=4096, N=1024, K=1024)
  gemm_bf16_tn<false><<<dim3(1024 / 128, 4096 / 256), 256, 0, stream>>>(
      attnb, wout, out, 4096, 1024, 1024);
}
